// SDPA_39668317946372
// MI455X (gfx1250) — matrix operand order repaired and checked by reading, not compiled, not run on hardware
//
#include <hip/hip_runtime.h>
#include <hip/hip_bf16.h>

typedef __attribute__((ext_vector_type(16))) _Float16 v16h;
typedef __attribute__((ext_vector_type(8)))  _Float16 v8h;
typedef __attribute__((ext_vector_type(4)))  _Float16 v4h;
typedef __attribute__((ext_vector_type(8)))  float    v8f;
typedef __attribute__((ext_vector_type(4)))  float    v4f;

#define B_    4
#define HQ_   32
#define HKV_  8
#define SLEN  1024
#define D_    128
#define KSTR  136   // 128 + 8 f16 pad (row stride of K tile in LDS)
#define VSTR  40    // 32 + 8 f16 pad (row stride of transposed V tile)
#define PSTR  40    // 32 + 8 f16 pad (row stride of P tile)

__device__ __forceinline__ v8f wmma16(v16h a, v16h b, v8f c) {
  // v_wmma_f32_16x16x32_f16: (neg_a, A, neg_b, B, c_mod, C, reuse_a, reuse_b)
  return __builtin_amdgcn_wmma_f32_16x16x32_f16(false, a, false, b, (short)0, c,
                                                false, false);
}

__global__ __launch_bounds__(128)
void sdpa_flash_wmma(const float* __restrict__ Q, const float* __restrict__ K,
                     const float* __restrict__ V, const int* __restrict__ is_causal,
                     float* __restrict__ O) {
  // double-buffered K/V staging tiles (ping-pong)
  __shared__ __align__(16) _Float16 Kl[2][32 * KSTR];     // [buf][key][d] f16
  __shared__ __align__(16) _Float16 Vl[2][D_ * VSTR];     // [buf][d][key] f16 (transposed)
  __shared__ __align__(16) _Float16 Pl[4 * 16 * PSTR];    // per-wave P tile

  const int tid  = threadIdx.x;
  const int lane = tid & 31;
  const int w    = tid >> 5;      // wave id 0..3
  const int l16  = lane & 15;
  const int half = lane >> 4;     // lane half (0/1)

  const int bx = blockIdx.x;
  const int qt = bx & 15;         // 16 q-tiles of 64 rows
  const int h  = (bx >> 4) & 31;  // HQ
  const int b  = bx >> 9;
  const int q_block = qt * 64;
  const int q0 = q_block + w * 16;            // this wave's first query row
  const int causal = is_causal[0];

  const float scale = 0.08838834764831845f;   // 1/sqrt(128)

  const int hk = h >> 2;  // GQA: HQ/HKV = 4
  const float* kg = K + (size_t)(b * HKV_ + hk) * SLEN * D_;
  const float* vg = V + (size_t)(b * HKV_ + hk) * SLEN * D_;

  // Per-thread staging coordinates: thread handles 8 float4 chunks of the
  // 32x128 tile: chunk lin = tid + 128*i -> key = lin/32, d4 = lin%32.
  const int nsteps = causal ? ((q_block >> 5) + 2) : (SLEN / 32);

  // ---- stage helper (called twice below) ----
  auto stage = [&](int kbase, int buf) {
    #pragma unroll
    for (int i = 0; i < 8; ++i) {
      const int lin = tid + 128 * i;
      const int key = lin >> 5;
      const int d4  = lin & 31;
      const size_t goff = (size_t)(kbase + key) * D_ + d4 * 4;
      v4f kq = *(const v4f*)(kg + goff);
      v4h kh;
      #pragma unroll
      for (int j = 0; j < 4; ++j) kh[j] = (_Float16)kq[j];
      *(v4h*)&Kl[buf][key * KSTR + d4 * 4] = kh;
      v4f vq = *(const v4f*)(vg + goff);
      #pragma unroll
      for (int j = 0; j < 4; ++j)
        Vl[buf][(d4 * 4 + j) * VSTR + key] = (_Float16)vq[j];
    }
  };

  // ---- Load Q tile (16 x 128) into A-fragments, scale folded, f32->f16 ----
  // A 16x32 layout: lane row M=l16; half0 holds K {0..15}, half1 {16..31}
  const size_t q_row_base = ((size_t)(b * HQ_ + h) * SLEN + (q0 + l16)) * D_;
  v16h qa[4];
  #pragma unroll
  for (int c = 0; c < 4; ++c) {
    const float* p0 = Q + q_row_base + 32 * c + 16 * half;
    const float* p1 = Q + q_row_base + 32 * c + 16 * half + 8;
    v4f t[4];
    t[0] = *(const v4f*)(p0);     t[1] = *(const v4f*)(p0 + 4);
    t[2] = *(const v4f*)(p1);     t[3] = *(const v4f*)(p1 + 4);
    #pragma unroll
    for (int i = 0; i < 16; ++i)
      qa[c][i] = (_Float16)(t[i >> 2][i & 3] * scale);
  }

  // ---- flash-attention state ----
  v8f o[8];
  #pragma unroll
  for (int t = 0; t < 8; ++t) o[t] = {};
  float m_run[8], l_run[8];
  #pragma unroll
  for (int r = 0; r < 8; ++r) { m_run[r] = -3.0e38f; l_run[r] = 0.0f; }

  // prologue: fill buffer 0
  stage(0, 0);
  __syncthreads();

  int buf = 0;
  for (int step = 0; step < nsteps; ++step) {
    const int kbase = step * 32;

    // prefetch next tile into the other buffer (overlaps with compute below)
    if (step + 1 < nsteps) stage(kbase + 32, buf ^ 1);

    const bool compute = (!causal) || (kbase <= q0 + 15);   // wave-uniform
    if (compute) {
      // ---- S = Q * K^T  (two 16x16 N-tiles, K-dim = 128 in 4 chunks) ----
      v8f s0 = {}, s1 = {};
      #pragma unroll
      for (int c = 0; c < 4; ++c) {
        // B 32x16 layout: lane col N=l16 (=key), half g holds d rows 16g..16g+15
        const _Float16* k0 = &Kl[buf][(l16)      * KSTR + 32 * c + 16 * half];
        const _Float16* k1 = &Kl[buf][(16 + l16) * KSTR + 32 * c + 16 * half];
        v8h a0 = *(const v8h*)k0, a1 = *(const v8h*)(k0 + 8);
        v8h b0 = *(const v8h*)k1, b1 = *(const v8h*)(k1 + 8);
        v16h kb0, kb1;
        #pragma unroll
        for (int i = 0; i < 8; ++i) {
          kb0[i] = a0[i]; kb0[i + 8] = a1[i];
          kb1[i] = b0[i]; kb1[i + 8] = b1[i];
        }
        s0 = wmma16(qa[c], kb0, s0);
        s1 = wmma16(qa[c], kb1, s1);
      }

      // ---- causal mask (diagonal step only) ----
      if (causal && (kbase + 31 > q0 + 15)) {
        #pragma unroll
        for (int r = 0; r < 8; ++r) {
          const int qi = q0 + r + 8 * half;
          if (kbase + l16      > qi) s0[r] = -3.0e38f;
          if (kbase + 16 + l16 > qi) s1[r] = -3.0e38f;
        }
      }

      // ---- online softmax (row stats across 16-lane half via shfl_xor) ----
      _Float16 pv0[8], pv1[8];
      #pragma unroll
      for (int r = 0; r < 8; ++r) {
        float mx = fmaxf(s0[r], s1[r]);
        #pragma unroll
        for (int off = 1; off < 16; off <<= 1)
          mx = fmaxf(mx, __shfl_xor(mx, off, 32));
        const float m_new = fmaxf(m_run[r], mx);
        const float e0 = __expf(s0[r] - m_new);
        const float e1 = __expf(s1[r] - m_new);
        float rs = e0 + e1;
        #pragma unroll
        for (int off = 1; off < 16; off <<= 1)
          rs += __shfl_xor(rs, off, 32);
        const float alpha = __expf(m_run[r] - m_new);
        l_run[r] = l_run[r] * alpha + rs;
        m_run[r] = m_new;
        #pragma unroll
        for (int t = 0; t < 8; ++t) o[t][r] *= alpha;
        pv0[r] = (_Float16)e0;
        pv1[r] = (_Float16)e1;
      }

      // ---- P: C-layout -> LDS -> A-layout (wave-private buffer) ----
      _Float16* pw = &Pl[w * 16 * PSTR];
      #pragma unroll
      for (int r = 0; r < 8; ++r) {
        const int M = r + 8 * half;
        pw[M * PSTR + l16]      = pv0[r];
        pw[M * PSTR + 16 + l16] = pv1[r];
      }
      const _Float16* pr = &Pl[w * 16 * PSTR + l16 * PSTR];
      v8h p0 = *(const v8h*)(pr + 16 * half);
      v8h p1 = *(const v8h*)(pr + 16 * half + 8);
      v16h pa;
      #pragma unroll
      for (int i = 0; i < 8; ++i) { pa[i] = p0[i]; pa[i + 8] = p1[i]; }

      // ---- O += P (16x32) * V (32x16 per d-tile), 8 d-tiles ----
      #pragma unroll
      for (int t = 0; t < 8; ++t) {
        const _Float16* vr = &Vl[buf][(16 * t + l16) * VSTR + 16 * half];
        v8h v0 = *(const v8h*)vr, v1 = *(const v8h*)(vr + 8);
        v16h vf;
        #pragma unroll
        for (int i = 0; i < 8; ++i) { vf[i] = v0[i]; vf[i + 8] = v1[i]; }
        o[t] = wmma16(pa, vf, o[t]);
      }
    }

    __syncthreads();   // next tile staged + current tile consumed
    buf ^= 1;
  }

  // ---- normalize and store O (f32) ----
  float* op = O + ((size_t)(b * HQ_ + h) * SLEN + q0) * D_;
  #pragma unroll
  for (int r = 0; r < 8; ++r) {
    const int M = r + 8 * half;
    const float inv = 1.0f / l_run[r];
    #pragma unroll
    for (int t = 0; t < 8; ++t)
      op[(size_t)M * D_ + 16 * t + l16] = o[t][r] * inv;
  }
}

extern "C" void kernel_launch(void* const* d_in, const int* in_sizes, int n_in,
                              void* d_out, int out_size, void* d_ws, size_t ws_size,
                              hipStream_t stream) {
  (void)in_sizes; (void)n_in; (void)out_size; (void)d_ws; (void)ws_size;
  const float* q = (const float*)d_in[0];
  const float* k = (const float*)d_in[1];
  const float* v = (const float*)d_in[2];
  const int*  ic = (const int*)d_in[3];
  float* out = (float*)d_out;

  const int grid = B_ * HQ_ * (SLEN / 64);  // 2048 blocks of 128 threads (4 waves)
  sdpa_flash_wmma<<<grid, 128, 0, stream>>>(q, k, v, ic, out);
}
